// GCONV_Model_RelPos_9363028705504
// MI455X (gfx1250) — matrix operand order repaired and checked by reading, not compiled, not run on hardware
//
#include <hip/hip_runtime.h>
#include <hip/hip_bf16.h>

typedef __attribute__((ext_vector_type(8)))  float  v8f;
typedef __attribute__((ext_vector_type(8)))  __bf16 v8bf;
typedef __attribute__((ext_vector_type(16))) __bf16 v16bf;
typedef __attribute__((ext_vector_type(4)))  int    v4i;

#define LDS_FENCE() asm volatile("s_wait_dscnt 0x0" ::: "memory")

#if __has_builtin(__builtin_amdgcn_global_load_async_to_lds_b128)
#define HAVE_ASYNC_LDS 1
typedef __attribute__((address_space(1))) v4i* gv4i_p;  // global int4*
typedef __attribute__((address_space(3))) v4i* lv4i_p;  // LDS int4*
#endif

// 16-byte global -> LDS copy; async (ASYNCcnt) on CDNA5, sync fallback.
__device__ __forceinline__ void async_cp128(__bf16* l, const __bf16* g) {
#ifdef HAVE_ASYNC_LDS
    __builtin_amdgcn_global_load_async_to_lds_b128(
        (gv4i_p)(unsigned long long)g,
        (lv4i_p)(unsigned)(unsigned long long)l,
        0, 0);
#else
    *(uint4*)l = *(const uint4*)g;
#endif
}

template <int N>
__device__ __forceinline__ void wait_async() {
#ifdef HAVE_ASYNC_LDS
#if __has_builtin(__builtin_amdgcn_s_wait_asynccnt)
    __builtin_amdgcn_s_wait_asynccnt(N);
#else
    asm volatile("s_wait_asynccnt %0" ::"i"(N) : "memory");
#endif
#endif
}

// ---------------------------------------------------------------------------
// WMMA fragment loaders (wave32; A and B both hold contiguous k = 16*(lane/16) + s)
// ---------------------------------------------------------------------------
__device__ __forceinline__ v16bf ldA(const __bf16* t, int stride, int lane, int kc) {
    int row = lane & 15;
    int kb  = kc * 32 + ((lane & 16) ? 16 : 0);
    const __bf16* p = t + row * stride + kb;
    v8bf lo = *(const v8bf*)p;
    v8bf hi = *(const v8bf*)(p + 8);
    v16bf r;
#pragma unroll
    for (int i = 0; i < 8; ++i) { r[i] = lo[i]; r[i + 8] = hi[i]; }
    return r;
}

__device__ __forceinline__ v16bf ldB(const __bf16* w, int stride, int lane, int nt, int kc) {
    int n  = nt * 16 + (lane & 15);
    int kb = kc * 32 + ((lane & 16) ? 16 : 0);
    return *(const v16bf*)(w + n * stride + kb);
}

// ---------------------------------------------------------------------------
// Per-wave 2-layer MLP on a 16x128 bf16 tile. L2 weights/biases preloaded in
// registers: L2 (8 WMMAs) runs with zero LDS traffic.
// ---------------------------------------------------------------------------
__device__ __forceinline__ void mlp_tile(const __bf16* sT, __bf16* sH,
                                         const __bf16* sW1,
                                         const v16bf (&bw2)[2][4],
                                         const float (&bias1)[4],
                                         const float (&bias2)[4],
                                         int lane, v8f acc2[4]) {
    v8f acc1[4];
#pragma unroll
    for (int nt = 0; nt < 4; ++nt)
#pragma unroll
        for (int i = 0; i < 8; ++i) acc1[nt][i] = 0.0f;

#pragma unroll
    for (int kc = 0; kc < 4; ++kc) {
        v16bf a  = ldA(sT, 128, lane, kc);
        v16bf w0 = ldB(sW1, 128, lane, 0, kc);
        v16bf w1 = ldB(sW1, 128, lane, 1, kc);
        v16bf w2 = ldB(sW1, 128, lane, 2, kc);
        v16bf w3 = ldB(sW1, 128, lane, 3, kc);
        acc1[0] = __builtin_amdgcn_wmma_f32_16x16x32_bf16(false, a, false, w0, (short)0, acc1[0], false, false);
        acc1[1] = __builtin_amdgcn_wmma_f32_16x16x32_bf16(false, a, false, w1, (short)0, acc1[1], false, false);
        acc1[2] = __builtin_amdgcn_wmma_f32_16x16x32_bf16(false, a, false, w2, (short)0, acc1[2], false, false);
        acc1[3] = __builtin_amdgcn_wmma_f32_16x16x32_bf16(false, a, false, w3, (short)0, acc1[3], false, false);
    }

    const int n0 = lane & 15;
    const int mo = (lane & 16) ? 8 : 0;
    LDS_FENCE();
    // bias + ReLU + bf16 downconvert, re-stage as L2 A-tile (16x64 row-major)
#pragma unroll
    for (int nt = 0; nt < 4; ++nt) {
#pragma unroll
        for (int v = 0; v < 8; ++v) {
            float val = fmaxf(acc1[nt][v] + bias1[nt], 0.0f);
            sH[(v + mo) * 64 + nt * 16 + n0] = (__bf16)val;
        }
    }
    LDS_FENCE();

#pragma unroll
    for (int nt = 0; nt < 4; ++nt)
#pragma unroll
        for (int i = 0; i < 8; ++i) acc2[nt][i] = 0.0f;

#pragma unroll
    for (int kc = 0; kc < 2; ++kc) {
        v16bf a = ldA(sH, 64, lane, kc);
#pragma unroll
        for (int nt = 0; nt < 4; ++nt)
            acc2[nt] = __builtin_amdgcn_wmma_f32_16x16x32_bf16(
                false, a, false, bw2[kc][nt], (short)0, acc2[nt], false, false);
    }
#pragma unroll
    for (int nt = 0; nt < 4; ++nt)
#pragma unroll
        for (int v = 0; v < 8; ++v) acc2[nt][v] += bias2[nt];
}

// ---------------------------------------------------------------------------
// msg kernel: 4 waves/block, per-wave tile loop, double-buffered async staging.
// Stage = 8 full-EXEC async b128 ops: lane pairs, even lane -> h[src] half-row,
// odd lane -> e half-row. asynccnt<=8 ==> current buffer complete (in-order).
// ---------------------------------------------------------------------------
#define WV 4

__device__ __forceinline__ void stage_msg(const __bf16* __restrict__ h,
                                          const __bf16* __restrict__ e,
                                          const int* __restrict__ srcI,
                                          const int* __restrict__ dstI,
                                          int tile, int lane,
                                          __bf16* tb, int* dslot) {
    int  row  = lane >> 1;
    bool left = !(lane & 1);
    int  edge = tile * 16 + row;
    const __bf16* g = left ? (h + (size_t)srcI[edge] * 64)
                           : (e + (size_t)edge * 64);
    __bf16* l = tb + row * 128 + (left ? 0 : 64);
#pragma unroll
    for (int i = 0; i < 8; ++i) async_cp128(l + i * 8, g + i * 8);
    if (left) dslot[row] = dstI[edge];
}

__global__ __launch_bounds__(128)
void k_msg(const __bf16* __restrict__ h, const __bf16* __restrict__ e,
           const int* __restrict__ srcI, const int* __restrict__ dstI,
           const __bf16* __restrict__ gW1, const __bf16* __restrict__ gW2,
           const float* __restrict__ b1, const float* __restrict__ b2,
           float* __restrict__ aggr, int ntiles) {
    __shared__ __bf16 sW1[64 * 128];
    __shared__ __bf16 sW2[64 * 64];
    __shared__ __bf16 sA[WV][2][16 * 128];
    __shared__ int    sDst[WV][2][16];

    {   // stage weights once per block (16KB + 8KB)
        uint4* d1 = (uint4*)sW1; const uint4* s1 = (const uint4*)gW1;
        for (int i = threadIdx.x; i < (64 * 128 * 2) / 16; i += 128) d1[i] = s1[i];
        uint4* d2 = (uint4*)sW2; const uint4* s2 = (const uint4*)gW2;
        for (int i = threadIdx.x; i < (64 * 64 * 2) / 16; i += 128) d2[i] = s2[i];
    }
    __syncthreads();

    const int wave = threadIdx.x >> 5;
    const int lane = threadIdx.x & 31;
    const int n0   = lane & 15;
    const int mo   = (lane & 16) ? 8 : 0;

    // preload L2 weights + biases into registers (loop-invariant)
    v16bf bw2[2][4];
#pragma unroll
    for (int kc = 0; kc < 2; ++kc)
#pragma unroll
        for (int nt = 0; nt < 4; ++nt) bw2[kc][nt] = ldB(sW2, 64, lane, nt, kc);
    float bias1[4], bias2[4];
#pragma unroll
    for (int nt = 0; nt < 4; ++nt) { bias1[nt] = b1[nt * 16 + n0]; bias2[nt] = b2[nt * 16 + n0]; }

    const int tstep = (int)gridDim.x * WV;
    int tile = (int)blockIdx.x * WV + wave;
    if (tile >= ntiles) return;       // wave-uniform: full EXEC below

    int par = 0;
    stage_msg(h, e, srcI, dstI, tile, lane, &sA[wave][0][0], sDst[wave][0]);

    for (;;) {
        int nxt = tile + tstep;
        if (nxt < ntiles) {
            stage_msg(h, e, srcI, dstI, nxt, lane, &sA[wave][par ^ 1][0], sDst[wave][par ^ 1]);
            wait_async<8>();          // current tile's 8 async ops complete
        } else {
            wait_async<0>();
        }
        LDS_FENCE();

        v8f acc2[4];
        mlp_tile(&sA[wave][par][0], &sA[wave][par][0], sW1, bw2, bias1, bias2, lane, acc2);

#pragma unroll
        for (int v = 0; v < 8; ++v) {
            int node = sDst[wave][par][v + mo];
            float* ap = aggr + (size_t)node * 64 + n0;
#pragma unroll
            for (int nt = 0; nt < 4; ++nt)
                atomicAdd(ap + nt * 16, acc2[nt][v]);
        }

        if (nxt >= ntiles) break;
        tile = nxt;
        par ^= 1;
    }
}

// ---------------------------------------------------------------------------
// upd kernel: per 16-node tile, A = [h | aggr/deg], out -> h_out (bf16)
// ---------------------------------------------------------------------------
__global__ __launch_bounds__(256)
void k_upd(const __bf16* __restrict__ h_in, const float* __restrict__ aggr,
           const float* __restrict__ deg,
           const __bf16* __restrict__ gW1, const __bf16* __restrict__ gW2,
           const float* __restrict__ b1, const float* __restrict__ b2,
           __bf16* __restrict__ h_out, int ntiles) {
    __shared__ __bf16 sW1[64 * 128];
    __shared__ __bf16 sW2[64 * 64];
    __shared__ __bf16 sA[8][16 * 128];

    {
        uint4* d1 = (uint4*)sW1; const uint4* s1 = (const uint4*)gW1;
        for (int i = threadIdx.x; i < (64 * 128 * 2) / 16; i += 256) d1[i] = s1[i];
        uint4* d2 = (uint4*)sW2; const uint4* s2 = (const uint4*)gW2;
        for (int i = threadIdx.x; i < (64 * 64 * 2) / 16; i += 256) d2[i] = s2[i];
    }
    __syncthreads();

    const int wave = threadIdx.x >> 5;
    const int lane = threadIdx.x & 31;
    const int n0   = lane & 15;
    const int mo   = (lane & 16) ? 8 : 0;
    const int tile = (int)blockIdx.x * 8 + wave;
    if (tile >= ntiles) return;

    v16bf bw2[2][4];
#pragma unroll
    for (int kc = 0; kc < 2; ++kc)
#pragma unroll
        for (int nt = 0; nt < 4; ++nt) bw2[kc][nt] = ldB(sW2, 64, lane, nt, kc);
    float bias1[4], bias2[4];
#pragma unroll
    for (int nt = 0; nt < 4; ++nt) { bias1[nt] = b1[nt * 16 + n0]; bias2[nt] = b2[nt * 16 + n0]; }

    __bf16* sT = &sA[wave][0];
    const int r    = lane & 15;
    const int node = tile * 16 + r;
    if (lane < 16) {            // left half: h (already bf16)
        const uint4* hp = (const uint4*)(h_in + (size_t)node * 64);
        uint4* tp = (uint4*)(sT + r * 128);
#pragma unroll
        for (int i = 0; i < 8; ++i) tp[i] = hp[i];
    } else {                    // right half: aggr * (1/max(deg,1)), f32 -> bf16
        float rd = 1.0f / fmaxf(deg[node], 1.0f);
        const float4* ap = (const float4*)(aggr + (size_t)node * 64);
        __bf16* tp = sT + r * 128 + 64;
#pragma unroll
        for (int i = 0; i < 16; ++i) {
            float4 v = ap[i];
            tp[i * 4 + 0] = (__bf16)(v.x * rd);
            tp[i * 4 + 1] = (__bf16)(v.y * rd);
            tp[i * 4 + 2] = (__bf16)(v.z * rd);
            tp[i * 4 + 3] = (__bf16)(v.w * rd);
        }
    }
    LDS_FENCE();

    v8f acc2[4];
    mlp_tile(sT, sT, sW1, bw2, bias1, bias2, lane, acc2);

#pragma unroll
    for (int v = 0; v < 8; ++v) {
        __bf16* op = h_out + (size_t)(tile * 16 + v + mo) * 64 + n0;
#pragma unroll
        for (int nt = 0; nt < 4; ++nt) op[nt * 16] = (__bf16)acc2[nt][v];
    }
}

// ---------------------------------------------------------------------------
// Small helper kernels
// ---------------------------------------------------------------------------
__global__ void k_cvt(const float* __restrict__ src, __bf16* __restrict__ dst,
                      int K, int N) {
    int i = blockIdx.x * blockDim.x + threadIdx.x;
    if (i < K * N) {
        int k = i / N, n = i - k * N;
        dst[n * K + k] = (__bf16)src[i];
    }
}

__global__ void k_zero(float* __restrict__ p, int n) {
    int i = blockIdx.x * blockDim.x + threadIdx.x;
    if (i < n) p[i] = 0.0f;
}

__global__ void k_deg(const int* __restrict__ dstI, float* __restrict__ deg, int E) {
    int i = blockIdx.x * blockDim.x + threadIdx.x;
    if (i < E) atomicAdd(deg + dstI[i], 1.0f);
}

__global__ __launch_bounds__(64)
void k_embed(const float* __restrict__ xin, int din,
             const float* __restrict__ W1, const float* __restrict__ b1,
             const float* __restrict__ W2, const float* __restrict__ b2,
             __bf16* __restrict__ out, int nrows) {
    __shared__ float sh[64];
    int row = blockIdx.x;
    if (row >= nrows) return;
    int j = threadIdx.x;
    float t = b1[j];
    for (int k = 0; k < din; ++k) t = fmaf(xin[(size_t)row * din + k], W1[k * 64 + j], t);
    sh[j] = fmaxf(t, 0.0f);
    __syncthreads();
    float v = b2[j];
#pragma unroll 8
    for (int k = 0; k < 64; ++k) v = fmaf(sh[k], W2[k * 64 + j], v);
    out[(size_t)row * 64 + j] = (__bf16)v;
}

__global__ __launch_bounds__(64)
void k_decoder(const __bf16* __restrict__ h,
               const float* __restrict__ W1, const float* __restrict__ b1,
               const float* __restrict__ W2, const float* __restrict__ b2,
               float* __restrict__ out, int nrows) {
    __shared__ float sh[64];
    int row = blockIdx.x;
    if (row >= nrows) return;
    int j = threadIdx.x;
    float t = b1[j];
#pragma unroll 8
    for (int k = 0; k < 64; ++k) t = fmaf((float)h[(size_t)row * 64 + k], W1[k * 64 + j], t);
    sh[j] = fmaxf(t, 0.0f);
    __syncthreads();
    if (j < 3) {
        float v = b2[j];
#pragma unroll 8
        for (int k = 0; k < 64; ++k) v = fmaf(sh[k], W2[k * 3 + j], v);
        out[(size_t)row * 3 + j] = v;
    }
}

// ---------------------------------------------------------------------------
// Host orchestration
// ---------------------------------------------------------------------------
static inline size_t align256(size_t x) { return (x + 255) & ~(size_t)255; }

extern "C" void kernel_launch(void* const* d_in, const int* in_sizes, int n_in,
                              void* d_out, int out_size, void* d_ws, size_t ws_size,
                              hipStream_t stream) {
    if (n_in < 39) return;
    // Flatten order (jax pytree: dict keys sorted):
    // 0:x 1:edge_attr 2:edge_index
    // 3..26: convs[c] {msg{W1,W2,b1,b2}, upd{W1,W2,b1,b2}} c=0..2
    // 27..30: decoder ; 31..34: edge_embed ; 35..38: node_embed
    const float* x  = (const float*)d_in[0];
    const float* ea = (const float*)d_in[1];
    const int*   ei = (const int*)d_in[2];
    const int N = in_sizes[0] / 7;
    const int E = in_sizes[1] / 4;
    const int* srcI = ei;
    const int* dstI = ei + E;

    size_t offH0   = 0;
    size_t offH1   = offH0 + align256((size_t)N * 64 * 2);
    size_t offE    = offH1 + align256((size_t)N * 64 * 2);
    size_t offAggr = offE + align256((size_t)E * 64 * 2);
    size_t offDeg  = offAggr + align256((size_t)N * 64 * 4);
    size_t offW    = offDeg + align256((size_t)N * 4);
    size_t need    = offW + align256((size_t)3 * 24576 * 2);
    if (ws_size < need) return;

    char* ws = (char*)d_ws;
    __bf16* h0   = (__bf16*)(ws + offH0);
    __bf16* h1   = (__bf16*)(ws + offH1);
    __bf16* ebuf = (__bf16*)(ws + offE);
    float*  aggr = (float*)(ws + offAggr);
    float*  deg  = (float*)(ws + offDeg);
    __bf16* wb   = (__bf16*)(ws + offW);

    for (int c = 0; c < 3; ++c) {
        const float* mW1 = (const float*)d_in[3 + c * 8 + 0];
        const float* mW2 = (const float*)d_in[3 + c * 8 + 1];
        const float* uW1 = (const float*)d_in[3 + c * 8 + 4];
        const float* uW2 = (const float*)d_in[3 + c * 8 + 5];
        __bf16* base = wb + (size_t)c * 24576;
        k_cvt<<<(128 * 64 + 255) / 256, 256, 0, stream>>>(mW1, base, 128, 64);
        k_cvt<<<(64 * 64 + 255) / 256, 256, 0, stream>>>(mW2, base + 8192, 64, 64);
        k_cvt<<<(128 * 64 + 255) / 256, 256, 0, stream>>>(uW1, base + 12288, 128, 64);
        k_cvt<<<(64 * 64 + 255) / 256, 256, 0, stream>>>(uW2, base + 20480, 64, 64);
    }

    k_zero<<<(N + 255) / 256, 256, 0, stream>>>(deg, N);
    k_embed<<<N, 64, 0, stream>>>(x, 7,
        (const float*)d_in[35], (const float*)d_in[37],
        (const float*)d_in[36], (const float*)d_in[38], h0, N);
    k_embed<<<E, 64, 0, stream>>>(ea, 4,
        (const float*)d_in[31], (const float*)d_in[33],
        (const float*)d_in[32], (const float*)d_in[34], ebuf, E);
    k_deg<<<(E + 255) / 256, 256, 0, stream>>>(dstI, deg, E);

    const int ntE = E / 16;   // 50000
    const int ntN = N / 16;   // 3125
    int mblocks = (ntE + WV - 1) / WV;
    if (mblocks > 2048) mblocks = 2048;

    for (int c = 0; c < 3; ++c) {
        __bf16* hin  = (c & 1) ? h1 : h0;
        __bf16* hout = (c & 1) ? h0 : h1;
        __bf16* base = wb + (size_t)c * 24576;
        const float* mb1 = (const float*)d_in[3 + c * 8 + 2];
        const float* mb2 = (const float*)d_in[3 + c * 8 + 3];
        const float* ub1 = (const float*)d_in[3 + c * 8 + 6];
        const float* ub2 = (const float*)d_in[3 + c * 8 + 7];

        k_zero<<<((N * 64) + 255) / 256, 256, 0, stream>>>(aggr, N * 64);
        k_msg<<<mblocks, 128, 0, stream>>>(hin, ebuf, srcI, dstI,
            base, base + 8192, mb1, mb2, aggr, ntE);
        k_upd<<<(ntN + 7) / 8, 256, 0, stream>>>(hin, aggr, deg,
            base + 12288, base + 20480, ub1, ub2, hout, ntN);
    }

    k_decoder<<<N, 64, 0, stream>>>(h1,
        (const float*)d_in[27], (const float*)d_in[29],
        (const float*)d_in[28], (const float*)d_in[30],
        (float*)d_out, N);
    (void)out_size;
}
